// EncoderSelfAttention_72069551227038
// MI455X (gfx1250) — matrix operand order repaired and checked by reading, not compiled, not run on hardware
//
#include <hip/hip_runtime.h>

// ---------------------------------------------------------------------------
// EncoderSelfAttention for MI455X (gfx1250): bf16 WMMA everywhere, async
// global->LDS double buffering, pre-transposed weights so all LDS staging is
// contiguous b128 traffic.
// ---------------------------------------------------------------------------

typedef __attribute__((ext_vector_type(16))) __bf16 bf16x16;
typedef __attribute__((ext_vector_type(8)))  __bf16 bf16x8;
typedef __attribute__((ext_vector_type(4)))  __bf16 bf16x4;
typedef __attribute__((ext_vector_type(8)))  float  f32x8;
typedef __attribute__((ext_vector_type(4)))  int    i32x4;

#define WMMA_BF16(a, b, c) \
  __builtin_amdgcn_wmma_f32_16x16x32_bf16(false, (a), false, (b), (short)0, (c), false, false)

constexpr int Bc  = 4;     // batch
constexpr int Sc  = 2048;  // sequence length
constexpr int Dc  = 1024;  // d_model
constexpr int Hc  = 16;    // heads
constexpr int HDc = 64;    // head dim
constexpr int Wc  = 1024;  // weight dim

#if __has_builtin(__builtin_amdgcn_global_load_async_to_lds_b128)
#define HAVE_ASYNC_LDS 1
#else
#define HAVE_ASYNC_LDS 0
#endif

// Copy 16 bytes global -> LDS. Async path uses the CDNA5 async LDS loads
// (ASYNCcnt); fallback is a plain b128 load + ds_store_b128.
__device__ __forceinline__ void cp16_g2l(const __bf16* g, __bf16* l) {
#if HAVE_ASYNC_LDS
  __builtin_amdgcn_global_load_async_to_lds_b128(
      (__attribute__((address_space(1))) i32x4*)g,
      (__attribute__((address_space(3))) i32x4*)l, 0, 0);
#else
  *(uint4*)l = *(const uint4*)g;
#endif
}

__device__ __forceinline__ void wait_async() {
#if HAVE_ASYNC_LDS
#if __has_builtin(__builtin_amdgcn_s_wait_asynccnt)
  __builtin_amdgcn_s_wait_asynccnt(0);
#else
  asm volatile("s_wait_asynccnt 0" ::: "memory");
#endif
#endif
}

// Build a 16-wide bf16 WMMA fragment from two contiguous 16B chunks
// (elements 0..7 from p0, 8..15 from p1) -> two ds_load_b128.
__device__ __forceinline__ bf16x16 ld_frag(const __bf16* p0, const __bf16* p1) {
  bf16x8 lo = *(const bf16x8*)p0;
  bf16x8 hi = *(const bf16x8*)p1;
  bf16x16 r;
#pragma unroll
  for (int i = 0; i < 8; ++i) { r[i] = lo[i]; r[i + 8] = hi[i]; }
  return r;
}

// ---------------------------------------------------------------------------
// fp32 -> bf16 conversion (vectorized x4)
// ---------------------------------------------------------------------------
__global__ void cvt_f32_bf16(const float* __restrict__ in,
                             __bf16* __restrict__ out, int n4) {
  int i = blockIdx.x * blockDim.x + threadIdx.x;
  if (i < n4) {
    float4 v = ((const float4*)in)[i];
    bf16x4 o;
    o[0] = (__bf16)v.x; o[1] = (__bf16)v.y;
    o[2] = (__bf16)v.z; o[3] = (__bf16)v.w;
    ((bf16x4*)out)[i] = o;
  }
}

// ---------------------------------------------------------------------------
// fp32 [K][N] -> bf16 [N][K] weight transpose (one-time, tiny)
// ---------------------------------------------------------------------------
__global__ __launch_bounds__(256) void transpose_f32_bf16(
    const float* __restrict__ in, __bf16* __restrict__ out) {
  __shared__ float tile[32][33];
  int k0 = blockIdx.y * 32, n0 = blockIdx.x * 32;
  int tx = threadIdx.x & 31, ty = threadIdx.x >> 5;  // 32 x 8
#pragma unroll
  for (int i = 0; i < 4; ++i)
    tile[ty + i * 8][tx] = in[(size_t)(k0 + ty + i * 8) * Wc + n0 + tx];
  __syncthreads();
#pragma unroll
  for (int i = 0; i < 4; ++i)
    out[(size_t)(n0 + ty + i * 8) * Dc + k0 + tx] = (__bf16)tile[tx][ty + i * 8];
}

// ---------------------------------------------------------------------------
// Stage a 128x32 bf16 tile (row stride 1024 in global) into LDS (stride 40).
// 256 threads x 2 chunks of 8 bf16.
// ---------------------------------------------------------------------------
__device__ __forceinline__ void stage_tile(const __bf16* __restrict__ gbase,
                                           __bf16* lbuf, int t) {
#pragma unroll
  for (int c0 = 0; c0 < 2; ++c0) {
    int c    = t * 2 + c0;           // 0..511
    int row  = c >> 2;               // 0..127
    int part = (c & 3) * 8;          // 0,8,16,24
    cp16_g2l(gbase + (size_t)row * 1024 + part, lbuf + row * 40 + part);
  }
}

// ---------------------------------------------------------------------------
// bf16 GEMM: C(MxN) = A(MxK) * Bt^T where Bt is [N][K] (pre-transposed).
// K = N = 1024. Block 256 threads (8 waves), tile 128x128, wave 32x64.
// Double-buffered LDS, async global->LDS copies overlapped with WMMA.
// MODE 0: fp32 row-major; 1: bf16 [B][H][S][hd]; 2: bf16 [B][H][hd][S]
// ---------------------------------------------------------------------------
template <int MODE>
__global__ __launch_bounds__(256) void gemm_bf16(const __bf16* __restrict__ A,
                                                 const __bf16* __restrict__ Bt,
                                                 void* __restrict__ Out) {
  constexpr int GK = 1024, GN = 1024;
  constexpr int LD = 40;                 // 80B row -> 16B aligned
  __shared__ __bf16 Al[2][128 * LD];
  __shared__ __bf16 Bl[2][128 * LD];

  const int t    = threadIdx.x;
  const int lane = t & 31;
  const int w    = t >> 5;
  const int wm   = w & 3, wn = w >> 2;   // 4x2 wave grid
  const int nl   = lane & 15;
  const int kh   = lane >> 4;
  const int tileM = blockIdx.y * 128;
  const int tileN = blockIdx.x * 128;

  const __bf16* Abase = A  + (size_t)tileM * GK;
  const __bf16* Bbase = Bt + (size_t)tileN * GK;

  f32x8 acc[2][4];
#pragma unroll
  for (int i = 0; i < 2; ++i)
#pragma unroll
    for (int j = 0; j < 4; ++j)
      acc[i][j] = (f32x8){0.f, 0.f, 0.f, 0.f, 0.f, 0.f, 0.f, 0.f};

  // loop-invariant LDS fragment offsets
  int aoff[2], boff[4];
#pragma unroll
  for (int i = 0; i < 2; ++i) aoff[i] = (wm * 32 + i * 16 + nl) * LD + kh * 8;
#pragma unroll
  for (int j = 0; j < 4; ++j) boff[j] = (wn * 64 + j * 16 + nl) * LD + kh * 8;

  stage_tile(Abase, Al[0], t);
  stage_tile(Bbase, Bl[0], t);
  wait_async();
  __syncthreads();

  int cur = 0;
  for (int kt = 0; kt < GK; kt += 32) {
    if (kt + 32 < GK) {                  // issue next tile while computing
      stage_tile(Abase + kt + 32, Al[cur ^ 1], t);
      stage_tile(Bbase + kt + 32, Bl[cur ^ 1], t);
    }
    bf16x16 af[2], bfg[4];
#pragma unroll
    for (int i = 0; i < 2; ++i)
      af[i] = ld_frag(&Al[cur][aoff[i]], &Al[cur][aoff[i] + 16]);
#pragma unroll
    for (int j = 0; j < 4; ++j)
      bfg[j] = ld_frag(&Bl[cur][boff[j]], &Bl[cur][boff[j] + 16]);
#pragma unroll
    for (int i = 0; i < 2; ++i)
#pragma unroll
      for (int j = 0; j < 4; ++j)
        acc[i][j] = WMMA_BF16(af[i], bfg[j], acc[i][j]);
    wait_async();
    __syncthreads();
    cur ^= 1;
  }

  // ---- epilogue (C/D layout: row = vgpr + 8*(lane>=16), col = lane&15) ----
  const int bIdx  = tileM >> 11;         // batch constant per block (S=2048)
  const int sBase = tileM & (Sc - 1);
#pragma unroll
  for (int i = 0; i < 2; ++i) {
    const int row0 = wm * 32 + i * 16 + 8 * kh;
#pragma unroll
    for (int j = 0; j < 4; ++j) {
      const int gcol = tileN + wn * 64 + j * 16 + nl;
      if (MODE == 0) {
        float* p = (float*)Out + (size_t)(tileM + row0) * GN + gcol;
#pragma unroll
        for (int r = 0; r < 8; ++r) p[(size_t)r * GN] = acc[i][j][r];
      } else if (MODE == 1) {            // Q/K: [B][H][S][hd]
        int h = gcol >> 6, d = gcol & (HDc - 1);
        __bf16* p = (__bf16*)Out +
                    (((size_t)bIdx * Hc + h) * Sc + sBase + row0) * HDc + d;
#pragma unroll
        for (int r = 0; r < 8; ++r) p[(size_t)r * HDc] = (__bf16)acc[i][j][r];
      } else {                           // V transposed: [B][H][hd][S]
        int h = gcol >> 6, d = gcol & (HDc - 1);
        __bf16* p = (__bf16*)Out +
                    (((size_t)bIdx * Hc + h) * HDc + d) * Sc + sBase + row0;
        bf16x8 pk;
#pragma unroll
        for (int r = 0; r < 8; ++r) pk[r] = (__bf16)acc[i][j][r];
        *(bf16x8*)p = pk;                // 8 consecutive seq positions
      }
    }
  }
}

// ---------------------------------------------------------------------------
// Flash attention: block = (b, h, 64-query tile); 128 threads = 4 waves,
// each wave owns 16 query rows. Key tiles of 64, double-buffered in LDS.
// ---------------------------------------------------------------------------
__global__ __launch_bounds__(128) void attn_kernel(const __bf16* __restrict__ Q,
                                                   const __bf16* __restrict__ K,
                                                   const __bf16* __restrict__ Vt,
                                                   const int* __restrict__ mask,
                                                   __bf16* __restrict__ Ctx) {
  constexpr int LT = 72;                 // 144B row -> 16B aligned
  __shared__ __bf16 Kl[2][64 * LT];      // [key][d]
  __shared__ __bf16 Vl[2][64 * LT];      // [d][key] (V pre-transposed)
  __shared__ __bf16 Pl[64 * LT];         // [q][key] probabilities
  __shared__ float  kml[2][64];
  __shared__ float  qml[64];

  const int t = threadIdx.x, lane = t & 31, w = t >> 5;
  const int nl = lane & 15, kh = lane >> 4;
  const int qt = blockIdx.x, h = blockIdx.y, b = blockIdx.z;
  const int qbase = qt * 64;
  const float SCALE = 0.03125f;          // 1/sqrt(1024)

  const __bf16* Qb = Q  + ((size_t)b * Hc + h) * Sc * HDc;
  const __bf16* Kb = K  + ((size_t)b * Hc + h) * Sc * HDc;
  const __bf16* Vb = Vt + ((size_t)b * Hc + h) * HDc * Sc;

  // Q fragments (A-layout: lane = M = q row; lane-half kh holds d 8*kh..+7, 16+8*kh..+7 per 32)
  bf16x16 aq[2];
  {
    const __bf16* qrow = Qb + (size_t)(qbase + w * 16 + nl) * HDc;
    int kb = kh * 8;
    aq[0] = ld_frag(qrow + kb,      qrow + 16 + kb);
    aq[1] = ld_frag(qrow + 32 + kb, qrow + 48 + kb);
  }

  // stage tile (K, V, key-mask) into a buffer
  auto stage_kv = [&](int kt, int buf) {
#pragma unroll
    for (int c0 = 0; c0 < 4; ++c0) {
      int c = t * 4 + c0;                // 512 chunks of 8 bf16
      int row = c >> 3, col = (c & 7) * 8;
      cp16_g2l(Kb + (size_t)(kt + row) * HDc + col, &Kl[buf][row * LT + col]);
      cp16_g2l(Vb + (size_t)row * Sc + kt + col,    &Vl[buf][row * LT + col]);
    }
    if (t < 64) kml[buf][t] = (float)mask[(size_t)b * Sc + kt + t];
  };

  if (t < 64) qml[t] = (float)mask[(size_t)b * Sc + qbase + t];
  stage_kv(0, 0);
  wait_async();
  __syncthreads();

  float qmv[8];
#pragma unroll
  for (int r = 0; r < 8; ++r) qmv[r] = qml[w * 16 + r + 8 * kh];

  f32x8 o[4];
#pragma unroll
  for (int j = 0; j < 4; ++j) o[j] = (f32x8){0.f,0.f,0.f,0.f,0.f,0.f,0.f,0.f};
  float m_s[8], l_s[8];
#pragma unroll
  for (int r = 0; r < 8; ++r) { m_s[r] = -3.0e38f; l_s[r] = 0.f; }

  int cur = 0;
  for (int kt = 0; kt < Sc; kt += 64) {
    if (kt + 64 < Sc) stage_kv(kt + 64, cur ^ 1);   // overlap with compute

    // ---- S = Q * K^T ----
    f32x8 sacc[4];
#pragma unroll
    for (int j = 0; j < 4; ++j) sacc[j] = (f32x8){0.f,0.f,0.f,0.f,0.f,0.f,0.f,0.f};
#pragma unroll
    for (int dk = 0; dk < 2; ++dk)
#pragma unroll
      for (int j = 0; j < 4; ++j) {
        const __bf16* kp = &Kl[cur][(j * 16 + nl) * LT + dk * 32 + kh * 8];
        bf16x16 bk = ld_frag(kp, kp + 16);
        sacc[j] = WMMA_BF16(aq[dk], bk, sacc[j]);
      }

    float kmlv[4];
#pragma unroll
    for (int j = 0; j < 4; ++j) kmlv[j] = kml[cur][j * 16 + nl];

    // ---- online softmax ----
#pragma unroll
    for (int r = 0; r < 8; ++r) {
      float tv[4], tmax = -3.0e38f;
#pragma unroll
      for (int j = 0; j < 4; ++j) {
        float x = (kmlv[j] != 0.f && qmv[r] != 0.f) ? sacc[j][r] * SCALE : -1.0e37f;
        tv[j] = x;
        tmax  = fmaxf(tmax, x);
      }
      for (int off = 8; off >= 1; off >>= 1)
        tmax = fmaxf(tmax, __shfl_xor(tmax, off, 32));
      float mnew = fmaxf(m_s[r], tmax);
      float corr = __expf(m_s[r] - mnew);
      float psum = 0.f;
      int prow = w * 16 + r + 8 * kh;
#pragma unroll
      for (int j = 0; j < 4; ++j) {
        float p = __expf(tv[j] - mnew);
        psum += p;
        Pl[prow * LT + j * 16 + nl] = (__bf16)p;
      }
      for (int off = 8; off >= 1; off >>= 1)
        psum += __shfl_xor(psum, off, 32);
      l_s[r] = l_s[r] * corr + psum;
      m_s[r] = mnew;
#pragma unroll
      for (int j = 0; j < 4; ++j) o[j][r] *= corr;
    }
    __syncthreads();                     // P visible

    // ---- O += P * V ----
#pragma unroll
    for (int dk = 0; dk < 2; ++dk) {
      const __bf16* pp = &Pl[(w * 16 + nl) * LT + dk * 32 + kh * 8];
      bf16x16 ap = ld_frag(pp, pp + 16);
#pragma unroll
      for (int j = 0; j < 4; ++j) {
        const __bf16* vp = &Vl[cur][(j * 16 + nl) * LT + dk * 32 + kh * 8];
        bf16x16 bv = ld_frag(vp, vp + 16);
        o[j] = WMMA_BF16(ap, bv, o[j]);
      }
    }
    wait_async();
    __syncthreads();
    cur ^= 1;
  }

  // ---- normalize + write ctx bf16 [B][S][H*hd] ----
#pragma unroll
  for (int r = 0; r < 8; ++r) {
    float inv = 1.0f / l_s[r];
    int q = qbase + w * 16 + r + 8 * kh;
#pragma unroll
    for (int j = 0; j < 4; ++j)
      Ctx[((size_t)b * Sc + q) * Wc + h * HDc + j * 16 + nl] =
          (__bf16)(o[j][r] * inv);
  }
}

// ---------------------------------------------------------------------------
extern "C" void kernel_launch(void* const* d_in, const int* in_sizes, int n_in,
                              void* d_out, int out_size, void* d_ws, size_t ws_size,
                              hipStream_t stream) {
  const float* x     = (const float*)d_in[0];
  const int*   amask = (const int*)d_in[1];
  const float* wq    = (const float*)d_in[2];
  const float* wk    = (const float*)d_in[3];
  const float* wv    = (const float*)d_in[4];
  const float* wo    = (const float*)d_in[5];

  const size_t nX = (size_t)Bc * Sc * Dc;     // 8M
  const size_t nW = (size_t)Dc * Wc;          // 1M
  const size_t nQ = (size_t)Bc * Hc * Sc * HDc;

  char* ws = (char*)d_ws;
  __bf16* Xb  = (__bf16*)(ws);                 size_t off = nX * 2;
  __bf16* Wqt = (__bf16*)(ws + off);           off += nW * 2;   // [N][K] bf16
  __bf16* Wkt = (__bf16*)(ws + off);           off += nW * 2;
  __bf16* Wvt = (__bf16*)(ws + off);           off += nW * 2;
  __bf16* Wot = (__bf16*)(ws + off);           off += nW * 2;
  __bf16* Qb  = (__bf16*)(ws + off);           off += nQ * 2;
  __bf16* Kb  = (__bf16*)(ws + off);           off += nQ * 2;
  __bf16* Vtb = (__bf16*)(ws + off);           off += nQ * 2;
  __bf16* Ctx = (__bf16*)(ws + off);           off += nX * 2;

  // activations: fp32 -> bf16; weights: fp32 [K][N] -> bf16 [N][K]
  cvt_f32_bf16<<<(int)(nX / 4 / 256), 256, 0, stream>>>(x, Xb, (int)(nX / 4));
  dim3 tgrid(Wc / 32, Dc / 32);
  transpose_f32_bf16<<<tgrid, 256, 0, stream>>>(wq, Wqt);
  transpose_f32_bf16<<<tgrid, 256, 0, stream>>>(wk, Wkt);
  transpose_f32_bf16<<<tgrid, 256, 0, stream>>>(wv, Wvt);
  transpose_f32_bf16<<<tgrid, 256, 0, stream>>>(wo, Wot);

  dim3 ggrid(Wc / 128, (Bc * Sc) / 128);       // 8 x 64
  gemm_bf16<1><<<ggrid, 256, 0, stream>>>(Xb, Wqt, (void*)Qb);
  gemm_bf16<1><<<ggrid, 256, 0, stream>>>(Xb, Wkt, (void*)Kb);
  gemm_bf16<2><<<ggrid, 256, 0, stream>>>(Xb, Wvt, (void*)Vtb);

  dim3 agrid(Sc / 64, Hc, Bc);                 // 32 x 16 x 4
  attn_kernel<<<agrid, 128, 0, stream>>>(Qb, Kb, Vtb, amask, Ctx);

  gemm_bf16<0><<<ggrid, 256, 0, stream>>>(Ctx, Wot, d_out);
}
